// WindowAttention_31748398252792
// MI455X (gfx1250) — matrix operand order repaired and checked by reading, not compiled, not run on hardware
//
#include <hip/hip_runtime.h>
#include <hip/hip_bf16.h>
#include <math.h>

typedef _Float16 h16;
typedef __attribute__((ext_vector_type(16))) _Float16 v16h;
typedef __attribute__((ext_vector_type(8)))  float    v8f;
typedef __attribute__((ext_vector_type(4)))  float    f32x4;

#define NTOK 49
#define DIM  128
#define HEADS 4
#define HD   32

union FragU { f32x4 f[2]; v16h h; };

// A fragment: 16x32 f16, row-major source with row stride S (halves).
// ISA layout: lanes 0-15 -> rows, K base 0; lanes 16-31 -> same rows, K base +8.
// elements 0..7 = K[base..base+7], elements 8..15 = K[base+16..base+23].
__device__ __forceinline__ v16h frag_a(const h16* A, int S, int mt, int k0, int lane) {
    int m  = mt * 16 + (lane & 15);
    int kb = k0 + ((lane >> 4) << 3);
    FragU u;
    u.f[0] = *(const f32x4*)(A + m * S + kb);
    u.f[1] = *(const f32x4*)(A + m * S + kb + 16);
    return u.h;
}

// B fragment: 32x16 (KxN). Source is B^T row-major (N rows x K cols), stride S.
// lanes 0-15 -> K base 0, lanes 16-31 -> K base +8; elements 0..7 = K[base..base+7], 8..15 = K[base+16..base+23].
__device__ __forceinline__ v16h frag_b(const h16* BT, int S, int nt, int k0, int lane) {
    int n  = nt * 16 + (lane & 15);
    int kb = k0 + ((lane >> 4) << 3);
    FragU u;
    const f32x4* p = (const f32x4*)(BT + n * S + kb);
    u.f[0] = p[0];
    u.f[1] = p[2];
    return u.h;
}

// Generic tiled GEMM over 8 waves: C(64 x NT*16) = A(64 x KD) * B(KD x NT*16)
template <int KD, int NT, int TT, typename StoreFn>
__device__ __forceinline__ void run_gemm(const h16* A, int As, const h16* BT, int Bs,
                                         int wave, int lane, StoreFn store) {
    constexpr int PER = TT / 8;
    #pragma unroll
    for (int t = wave * PER; t < wave * PER + PER; ++t) {
        int mt = t / NT, nt = t % NT;
        v8f acc = {};
        #pragma unroll
        for (int k = 0; k < KD; k += 32) {
            v16h a  = frag_a(A,  As, mt, k, lane);
            v16h bb = frag_b(BT, Bs, nt, k, lane);
            acc = __builtin_amdgcn_wmma_f32_16x16x32_f16(
                false, a, false, bb, (short)0, acc, false, false);
        }
        store(mt, nt, acc);
    }
}

__global__ __launch_bounds__(256)
void win_attn_kernel(const float* __restrict__ x1, const float* __restrict__ x2,
                     const float* __restrict__ x3, const float* __restrict__ mask,
                     const float* __restrict__ Wq, const float* __restrict__ bq,
                     const float* __restrict__ Wk, const float* __restrict__ bk,
                     const float* __restrict__ Wv, const float* __restrict__ bv,
                     const float* __restrict__ table,
                     const float* __restrict__ Wp, const float* __restrict__ bp,
                     float* __restrict__ outp, int nW) {
    __shared__ __align__(16) h16   w16[DIM * DIM];      // 32 KB: current weight (B^T layout = W row-major)
    __shared__ __align__(16) h16   xa16[64 * DIM];      // 16 KB: x1/x2/x3, later O
    __shared__ __align__(16) h16   q16[64 * DIM];       // 16 KB
    __shared__ __align__(16) h16   k16[64 * DIM];       // 16 KB (B^T for QK^T)
    __shared__ __align__(16) h16   vt16[DIM * 64];      // 16 KB: V transposed (channel-major)
    __shared__ __align__(16) float scoresS[64 * 64];    // 16 KB
    __shared__ __align__(16) h16   probs16[64 * 64];    //  8 KB (K padded to 64, tail zeroed)
    __shared__ float biasv[DIM];

    const int tid  = threadIdx.x;
    const int lane = tid & 31;
    const int wave = tid >> 5;
    const int b    = blockIdx.x;
    const float scale = 0.17677669529663687f;   // 1/sqrt(32)

    auto load_w = [&](const float* W) {
        for (int i = tid * 4; i < DIM * DIM; i += 1024) {
            f32x4 v = *(const f32x4*)(W + i);
            w16[i+0] = (h16)v[0]; w16[i+1] = (h16)v[1];
            w16[i+2] = (h16)v[2]; w16[i+3] = (h16)v[3];
        }
    };
    auto load_x = [&](const float* X) {
        for (int i = tid * 4; i < NTOK * DIM; i += 1024) {
            f32x4 v = *(const f32x4*)(X + i);
            xa16[i+0] = (h16)v[0]; xa16[i+1] = (h16)v[1];
            xa16[i+2] = (h16)v[2]; xa16[i+3] = (h16)v[3];
        }
        for (int i = NTOK * DIM + tid; i < 64 * DIM; i += 256) xa16[i] = (h16)0.f;
    };
    auto load_b = [&](const float* bb) { if (tid < DIM) biasv[tid] = bb[tid]; };

    auto st_rm = [&](h16* dst, int mt, int nt, v8f c) {   // row-major f16 + bias
        int n  = nt * 16 + (lane & 15);
        int m0 = mt * 16 + ((lane >> 4) << 3);
        float bb = biasv[n];
        #pragma unroll
        for (int i = 0; i < 8; ++i) dst[(m0 + i) * DIM + n] = (h16)(c[i] + bb);
    };

    // ---- Q = x1 @ Wq^T + bq -----------------------------------------------
    load_w(Wq); load_x(x1 + (size_t)b * NTOK * DIM); load_b(bq);
    __syncthreads();
    run_gemm<DIM, 8, 32>(xa16, DIM, w16, DIM, wave, lane,
        [&](int mt, int nt, v8f c) { st_rm(q16, mt, nt, c); });
    __syncthreads();

    // ---- K = x2 @ Wk^T + bk -----------------------------------------------
    load_w(Wk); load_x(x2 + (size_t)b * NTOK * DIM); load_b(bk);
    __syncthreads();
    run_gemm<DIM, 8, 32>(xa16, DIM, w16, DIM, wave, lane,
        [&](int mt, int nt, v8f c) { st_rm(k16, mt, nt, c); });
    __syncthreads();

    // ---- V = x3 @ Wv^T + bv, stored transposed (channel-major) ------------
    load_w(Wv); load_x(x3 + (size_t)b * NTOK * DIM); load_b(bv);
    __syncthreads();
    run_gemm<DIM, 8, 32>(xa16, DIM, w16, DIM, wave, lane,
        [&](int mt, int nt, v8f c) {
            int n  = nt * 16 + (lane & 15);              // channel
            int m0 = mt * 16 + ((lane >> 4) << 3);       // token base
            float bb = biasv[n];
            h16* p = &vt16[n * 64 + m0];                 // 8 contiguous halves
            #pragma unroll
            for (int i = 0; i < 8; ++i) p[i] = (h16)(c[i] + bb);
        });
    __syncthreads();

    // ---- per-head attention ----------------------------------------------
    const float* mrow_base = mask + (size_t)(b % nW) * NTOK * NTOK;
    for (int h = 0; h < HEADS; ++h) {
        // scores = q_h @ k_h^T  (K dim = 32)
        run_gemm<HD, 4, 16>(q16 + h * HD, DIM, k16 + h * HD, DIM, wave, lane,
            [&](int mt, int nt, v8f c) {
                int n  = nt * 16 + (lane & 15);
                int m0 = mt * 16 + ((lane >> 4) << 3);
                #pragma unroll
                for (int i = 0; i < 8; ++i) scoresS[(m0 + i) * 64 + n] = c[i];
            });
        __syncthreads();

        // softmax row-wise with scale + relative-position bias + mask (fp32)
        if (tid < NTOK) {
            int i = tid, ih = i / 7, iw = i % 7;
            const float* mrow = mrow_base + i * NTOK;
            float mx = -1e30f;
            for (int j = 0; j < NTOK; ++j) {
                int jh = j / 7, jw = j % 7;
                int idx = (ih - jh + 6) * 13 + (iw - jw + 6);
                float t = scoresS[i * 64 + j] * scale + table[idx * HEADS + h] + mrow[j];
                scoresS[i * 64 + j] = t;
                mx = fmaxf(mx, t);
            }
            float s = 0.f;
            for (int j = 0; j < NTOK; ++j) {
                float e = __expf(scoresS[i * 64 + j] - mx);
                scoresS[i * 64 + j] = e;
                s += e;
            }
            float inv = 1.f / s;
            for (int j = 0; j < NTOK; ++j)
                probs16[i * 64 + j] = (h16)(scoresS[i * 64 + j] * inv);
            for (int j = NTOK; j < 64; ++j) probs16[i * 64 + j] = (h16)0.f;
        }
        __syncthreads();

        // O_h = P @ V_h  (K padded to 64); write into xa16 (x3 no longer needed)
        run_gemm<64, 2, 8>(probs16, 64, vt16 + h * HD * 64, 64, wave, lane,
            [&](int mt, int nt, v8f c) {
                int n  = h * HD + nt * 16 + (lane & 15);
                int m0 = mt * 16 + ((lane >> 4) << 3);
                #pragma unroll
                for (int i = 0; i < 8; ++i) xa16[(m0 + i) * DIM + n] = (h16)c[i];
            });
        __syncthreads();
    }

    // ---- Y = O @ Wp^T + bp  -> global --------------------------------------
    load_w(Wp); load_b(bp);
    __syncthreads();
    run_gemm<DIM, 8, 32>(xa16, DIM, w16, DIM, wave, lane,
        [&](int mt, int nt, v8f c) {
            int n  = nt * 16 + (lane & 15);
            int m0 = mt * 16 + ((lane >> 4) << 3);
            float bb = biasv[n];
            #pragma unroll
            for (int i = 0; i < 8; ++i) {
                int m = m0 + i;
                if (m < NTOK)
                    outp[(size_t)b * NTOK * DIM + m * DIM + n] = c[i] + bb;
            }
        });
}

extern "C" void kernel_launch(void* const* d_in, const int* in_sizes, int n_in,
                              void* d_out, int out_size, void* d_ws, size_t ws_size,
                              hipStream_t stream) {
    const float* x1    = (const float*)d_in[0];
    const float* x2    = (const float*)d_in[1];
    const float* x3    = (const float*)d_in[2];
    const float* mask  = (const float*)d_in[3];
    const float* Wq    = (const float*)d_in[4];
    const float* bq    = (const float*)d_in[5];
    const float* Wk    = (const float*)d_in[6];
    const float* bk    = (const float*)d_in[7];
    const float* Wv    = (const float*)d_in[8];
    const float* bv    = (const float*)d_in[9];
    const float* table = (const float*)d_in[10];
    const float* Wp    = (const float*)d_in[11];
    const float* bp    = (const float*)d_in[12];
    float* out = (float*)d_out;

    int B_ = in_sizes[0] / (NTOK * DIM);
    int nW = in_sizes[3] / (NTOK * NTOK);

    win_attn_kernel<<<B_, 256, 0, stream>>>(x1, x2, x3, mask, Wq, bq, Wk, bk,
                                            Wv, bv, table, Wp, bp, out, nW);
}
